// ApertureAwareAttention_86878598464248
// MI455X (gfx1250) — matrix operand order repaired and checked by reading, not compiled, not run on hardware
//
#include <hip/hip_runtime.h>
#include <hip/hip_bf16.h>
#include <cstdint>

typedef __attribute__((ext_vector_type(16))) _Float16 v16h;
typedef __attribute__((ext_vector_type(8)))  float    v8f;
typedef __attribute__((ext_vector_type(4)))  int      v4i;
typedef __attribute__((ext_vector_type(4)))  unsigned int u32x4;
typedef __attribute__((ext_vector_type(8)))  int      i32x8;
typedef __attribute__((ext_vector_type(4)))  int      i32x4;

#define DEVINL __device__ __forceinline__

union H4 { _Float16 h[4]; uint2 u; };

// ---- optional CDNA5 async global->LDS copy (ASYNCcnt path) ----
#if defined(__has_builtin)
# if __has_builtin(__builtin_amdgcn_global_load_async_to_lds_b128)
#  define HAVE_ASYNC_LDS 1
# endif
# if __has_builtin(__builtin_amdgcn_tensor_load_to_lds)
#  define HAVE_TDM 1
# endif
#endif
#ifndef HAVE_ASYNC_LDS
# define HAVE_ASYNC_LDS 0
#endif
#ifndef HAVE_TDM
# define HAVE_TDM 0
#endif

#if HAVE_ASYNC_LDS
typedef __attribute__((address_space(1))) v4i gv4i_t;   // global int4
typedef __attribute__((address_space(3))) v4i lv4i_t;   // LDS int4
DEVINL void async_copy16(const void* g, void* l){
  __builtin_amdgcn_global_load_async_to_lds_b128(
      (gv4i_t*)(uintptr_t)g, (lv4i_t*)(uint32_t)(uintptr_t)l, 0, 0);
}
DEVINL void async_wait0(){ asm volatile("s_wait_asynccnt 0" ::: "memory"); }
#endif

#if HAVE_TDM
DEVINL void tdm_wait0(){
#if __has_builtin(__builtin_amdgcn_s_wait_tensorcnt)
  __builtin_amdgcn_s_wait_tensorcnt(0);
#else
  asm volatile("s_wait_tensorcnt 0" ::: "memory");
#endif
}
// One 2D tile load via the Tensor Data Mover (CDNA5 ISA ch.8 D# layout).
// rows x cols_halves f16 tile, global row stride in halves; LDS pitch padded by
// pad_amount (code: DWORDs-1) every pad_interval (code: log2(DWORDs)-1) DWORDs.
DEVINL void tdm_load_2d(const void* gsrc, void* lds_dst,
                        uint32_t rows, uint32_t cols_halves, uint64_t row_stride_halves,
                        uint32_t pad_interval_code, uint32_t pad_amount_code)
{
  uint64_t ga = (uint64_t)(uintptr_t)gsrc;
  uint32_t la = (uint32_t)(uintptr_t)lds_dst;
  u32x4 g0;
  g0[0] = 1u;                                        // count=1 (valid), user mode
  g0[1] = la;                                        // lds_addr [63:32]
  g0[2] = (uint32_t)(ga & 0xffffffffu);              // global_addr [95:64]
  g0[3] = (uint32_t)((ga >> 32) & 0x01ffffffu)       // global_addr [120:96]
        | (2u << 30);                                // type=2 ("image") [127:126]
  i32x8 g1;
  g1[0] = (int)((1u << 16)                           // data_size=1 -> 2 bytes
        | (1u << 20)                                 // pad_enable
        | (pad_interval_code << 22)
        | (pad_amount_code   << 25));
  g1[1] = (int)((cols_halves & 0xffffu) << 16);      // tensor_dim0[15:0] @ bit48
  g1[2] = (int)((cols_halves >> 16) & 0xffffu)       // tensor_dim0[31:16]
        | (int)((rows & 0xffffu) << 16);             // tensor_dim1[15:0] @ bit80
  g1[3] = (int)((rows >> 16) & 0xffffu)              // tensor_dim1[31:16]
        | (int)((cols_halves & 0xffffu) << 16);      // tile_dim0 @ bit112
  g1[4] = (int)(rows & 0xffffu);                     // tile_dim1 @ bit128 (tile_dim2=0)
  g1[5] = (int)(uint32_t)(row_stride_halves & 0xffffffffu);   // dim0_stride[31:0] @ bit160
  g1[6] = (int)(uint32_t)((row_stride_halves >> 32) & 0xffffu); // dim0_stride[47:32]
  g1[7] = 0;                                         // dim1_stride (unused, 2D)
  i32x4 gz; gz[0]=0; gz[1]=0; gz[2]=0; gz[3]=0;
  i32x8 gz8;
#pragma unroll
  for (int i=0;i<8;i++) gz8[i]=0;
  __builtin_amdgcn_tensor_load_to_lds(g0, g1, gz, gz, gz8, 0);
}
#endif

DEVINL v8f zero8(){
  v8f z;
#pragma unroll
  for (int i=0;i<8;i++) z[i]=0.f;
  return z;
}

DEVINL v8f wmma_f16(v16h a, v16h b, v8f c){
  // v_wmma_f32_16x16x32_f16: D = A(16x32) * B(32x16) + C
  return __builtin_amdgcn_wmma_f32_16x16x32_f16(false, a, false, b, (short)0, c, false, false);
}

// ---- WMMA fragment loaders (per CDNA5 ISA 7.12.2 layouts, wave32) ----

// A fragment (16x32 f16) from row-major [m][k] tile.
DEVINL v16h load_A(const _Float16* p, int stride, int row0, int k0, int lane){
  const _Float16* r = p + (size_t)(row0 + (lane & 15))*stride + k0 + (((lane>>4)&1)<<3);
  v16h a;
#pragma unroll
  for (int i=0;i<8;i++){
    int k = ((i&4)<<2) + ((i&3)<<1);   // 0,2,4,6,16,18,20,22
    a[2*i]   = r[k];
    a[2*i+1] = r[k+1];
  }
  return a;
}

// B fragment (32x16 f16), logical B[k][n] = src[n][k] (source row-major [n][k]),
// so each lane reads two contiguous 8-element K runs (same K order as load_A), no repacking.
DEVINL v16h load_BT(const _Float16* p, int stride, int col0, int k0, int lane){
  const _Float16* r = p + (size_t)(col0 + (lane & 15))*stride + k0 + (((lane>>4)&1)<<3);
  v16h b;
#pragma unroll
  for (int i=0;i<8;i++){ int k = ((i&4)<<2) + ((i&3)<<1); b[2*i]=r[k]; b[2*i+1]=r[k+1]; }
  return b;
}

// half-wave (16-lane group) reductions
DEVINL float redmax16(float v){
  v = fmaxf(v, __shfl_xor(v, 1, 32));
  v = fmaxf(v, __shfl_xor(v, 2, 32));
  v = fmaxf(v, __shfl_xor(v, 4, 32));
  v = fmaxf(v, __shfl_xor(v, 8, 32));
  return v;
}
DEVINL float redsum16(float v){
  v += __shfl_xor(v, 1, 32);
  v += __shfl_xor(v, 2, 32);
  v += __shfl_xor(v, 4, 32);
  v += __shfl_xor(v, 8, 32);
  return v;
}

// ---------------- fp32 -> f16 weight conversion WITH transpose ----------------
// in: W[k][n] fp32 (512x512). out: WT[n][k] f16.
__global__ void csa_cvtT_kernel(const float* __restrict__ s, _Float16* __restrict__ d){
  int gid = blockIdx.x*256 + threadIdx.x;   // 262144 total
  int n = gid & 511, k = gid >> 9;
  d[(size_t)n*512 + k] = (_Float16)s[(size_t)k*512 + n];
}

// ---------------- projection GEMM: Y = (X @ W + b) * alpha, f16 out ----------------
// X: (65536,512) fp32 row-major; WT: (512,512) f16 [n][k]; Y f16.
__global__ __launch_bounds__(256) void csa_proj_gemm_kernel(
    const float* __restrict__ X, const _Float16* __restrict__ WT,
    const float* __restrict__ bias, float alpha, _Float16* __restrict__ Y)
{
  const int tid = threadIdx.x, lane = tid & 31, wave = tid >> 5;
  const int m0 = blockIdx.x << 7, n0 = blockIdx.y << 7;
  __shared__ _Float16 Xs[128*40];   // [m][k] 128 x 32 (+8 pad)
  __shared__ _Float16 Wt[128*40];   // [n][k] 128 x 32 (+8 pad, 16B-aligned rows)
  v8f acc[8];
#pragma unroll
  for (int t=0;t<8;t++) acc[t] = zero8();

  for (int k0 = 0; k0 < 512; k0 += 32){
    __syncthreads();
#if HAVE_TDM
    if (tid < 32){                         // wave 0: TDM moves the whole W^T tile
      tdm_load_2d(WT + (size_t)n0*512 + k0, Wt, 128, 32, 512, /*16 dw*/3, /*4 dw*/3);
    }
#endif
#pragma unroll
    for (int j=0;j<4;j++){                 // stage X tile (fp32 -> f16)
      int i = tid + (j<<8);
      int r = i >> 3, c4 = (i & 7) << 2;
      float4 f = *(const float4*)(X + (size_t)(m0+r)*512 + k0 + c4);
      H4 hh;
      hh.h[0]=(_Float16)f.x; hh.h[1]=(_Float16)f.y;
      hh.h[2]=(_Float16)f.z; hh.h[3]=(_Float16)f.w;
      *(uint2*)&Xs[r*40 + c4] = hh.u;
    }
#if HAVE_TDM
    if (tid < 32) tdm_wait0();
#elif HAVE_ASYNC_LDS
#pragma unroll
    for (int j=0;j<2;j++){                 // stage W^T tile via async-to-LDS
      int i = tid + (j<<8);
      int r = i >> 2, c8 = (i & 3) << 3;
      async_copy16(WT + (size_t)(n0+r)*512 + k0 + c8, &Wt[r*40 + c8]);
    }
    async_wait0();
#else
#pragma unroll
    for (int j=0;j<4;j++){                 // stage W^T tile
      int i = tid + (j<<8);
      int r = i >> 3, c4 = (i & 7) << 2;
      *(uint2*)&Wt[r*40 + c4] = *(const uint2*)(WT + (size_t)(n0+r)*512 + k0 + c4);
    }
#endif
    if (k0 + 32 < 512){                    // prefetch next K tiles
      __builtin_prefetch(X  + (size_t)(m0 + ((tid*2)&127))*512 + k0 + 32, 0, 0);
      __builtin_prefetch(WT + (size_t)(n0 + ((tid*2)&127))*512 + k0 + 32, 0, 0);
    }
    __syncthreads();
    v16h a = load_A(Xs, 40, wave<<4, 0, lane);
#pragma unroll
    for (int t=0;t<8;t++)
      acc[t] = wmma_f16(a, load_BT(Wt, 40, t<<4, 0, lane), acc[t]);
  }

  const int hi = lane>>4, nn = lane & 15;
#pragma unroll
  for (int t=0;t<8;t++){
#pragma unroll
    for (int r=0;r<8;r++){
      int m = m0 + (wave<<4) + r + (hi<<3);
      int n = n0 + (t<<4) + nn;
      Y[(size_t)m*512 + n] = (_Float16)((acc[t][r] + bias[n]) * alpha);
    }
  }
}

// ---------------- output GEMM: out = (A1 + A2) @ Wo + bo, fp32 out ----------------
__global__ __launch_bounds__(256) void csa_out_gemm_kernel(
    const _Float16* __restrict__ A1, const _Float16* __restrict__ A2,
    const _Float16* __restrict__ WT, const float* __restrict__ bias,
    float* __restrict__ Y)
{
  const int tid = threadIdx.x, lane = tid & 31, wave = tid >> 5;
  const int m0 = blockIdx.x << 7, n0 = blockIdx.y << 7;
  __shared__ _Float16 Xs[128*40];
  __shared__ _Float16 Wt[128*40];
  v8f acc[8];
#pragma unroll
  for (int t=0;t<8;t++) acc[t] = zero8();

  for (int k0 = 0; k0 < 512; k0 += 32){
    __syncthreads();
#if HAVE_TDM
    if (tid < 32){
      tdm_load_2d(WT + (size_t)n0*512 + k0, Wt, 128, 32, 512, 3, 3);
    }
#endif
#pragma unroll
    for (int j=0;j<4;j++){                 // stage (A1 + A2) tile
      int i = tid + (j<<8);
      int r = i >> 3, c4 = (i & 7) << 2;
      size_t off = (size_t)(m0+r)*512 + k0 + c4;
      H4 a, b2, o;
      a.u  = *(const uint2*)(A1 + off);
      b2.u = *(const uint2*)(A2 + off);
#pragma unroll
      for (int q=0;q<4;q++) o.h[q] = (_Float16)((float)a.h[q] + (float)b2.h[q]);
      *(uint2*)&Xs[r*40 + c4] = o.u;
    }
#if HAVE_TDM
    if (tid < 32) tdm_wait0();
#elif HAVE_ASYNC_LDS
#pragma unroll
    for (int j=0;j<2;j++){
      int i = tid + (j<<8);
      int r = i >> 2, c8 = (i & 3) << 3;
      async_copy16(WT + (size_t)(n0+r)*512 + k0 + c8, &Wt[r*40 + c8]);
    }
    async_wait0();
#else
#pragma unroll
    for (int j=0;j<4;j++){
      int i = tid + (j<<8);
      int r = i >> 3, c4 = (i & 7) << 2;
      *(uint2*)&Wt[r*40 + c4] = *(const uint2*)(WT + (size_t)(n0+r)*512 + k0 + c4);
    }
#endif
    if (k0 + 32 < 512){
      __builtin_prefetch(A1 + (size_t)(m0 + ((tid*2)&127))*512 + k0 + 32, 0, 0);
      __builtin_prefetch(WT + (size_t)(n0 + ((tid*2)&127))*512 + k0 + 32, 0, 0);
    }
    __syncthreads();
    v16h a = load_A(Xs, 40, wave<<4, 0, lane);
#pragma unroll
    for (int t=0;t<8;t++)
      acc[t] = wmma_f16(a, load_BT(Wt, 40, t<<4, 0, lane), acc[t]);
  }

  const int hi = lane>>4, nn = lane & 15;
#pragma unroll
  for (int t=0;t<8;t++){
#pragma unroll
    for (int r=0;r<8;r++){
      int m = m0 + (wave<<4) + r + (hi<<3);
      int n = n0 + (t<<4) + nn;
      Y[(size_t)m*512 + n] = acc[t][r] + bias[n];
    }
  }
}

// ---------------- LePE depthwise 5x5 conv on V (f16 in/out) ----------------
__global__ void csa_lepe_kernel(const _Float16* __restrict__ v, const float* __restrict__ w,
                                const float* __restrict__ bias, _Float16* __restrict__ out)
{
  int gid = blockIdx.x*256 + threadIdx.x;      // ((b*128+y)*128+x)*512 + c
  int c = gid & 511;
  int pix = gid >> 9;
  int x = pix & 127, y = (pix >> 7) & 127, b = pix >> 14;
  float acc = bias[c];
#pragma unroll
  for (int ky=0; ky<5; ky++){
    int yy = y + ky - 2;
    if ((unsigned)yy >= 128u) continue;
#pragma unroll
    for (int kx=0; kx<5; kx++){
      int xx = x + kx - 2;
      if ((unsigned)xx >= 128u) continue;
      acc += (float)v[(((size_t)(b*128+yy)*128)+xx)*512 + c] * w[(ky*5+kx)*512 + c];
    }
  }
  out[(size_t)gid] = (_Float16)acc;
}

// ---------------- axial attention slice (128 tokens, head_dim 64) ----------------
// qp/kp/vp/op: element (row, d) at ptr + row*rstride + d.  mk: mask[q*128+k].
// smem (halves): [0..17407] = union{ Ks 128x72 | P 8x16x136 }, [17408..] = Vt 64x138
DEVINL void csa_attn_slice(const _Float16* qp, const _Float16* kp, const _Float16* vp,
                           const float* mk, _Float16* op, size_t rstride, _Float16* smem)
{
  const int tid = threadIdx.x, lane = tid & 31, wave = tid >> 5;
  _Float16* Ks = smem;            // [key][d]   128 x 72
  _Float16* P  = smem;            // probs, overlaid after barrier
  _Float16* Vt = smem + 17408;    // [d][key]   64 x 138 (transposed V)

  // stage K tile: TDM (one descriptor, pad 32dw interval / 4dw amount -> pitch 72)
#if HAVE_TDM
  if (tid < 32){
    tdm_load_2d(kp, Ks, 128, 64, (uint64_t)rstride, /*32 dw*/4, /*4 dw*/3);
  }
#elif HAVE_ASYNC_LDS
#pragma unroll
  for (int j=0;j<2;j++){
    int i = tid + (j<<8);
    int r = i >> 2, c8 = (i & 3) << 3;
    async_copy16(kp + (size_t)r*rstride + c8, &Ks[r*72 + c8]);
  }
#else
#pragma unroll
  for (int j=0;j<8;j++){
    int i = tid + (j<<8);
    int r = i >> 4, c4 = (i & 15) << 2;
    *(uint2*)&Ks[r*72 + c4] = *(const uint2*)(kp + (size_t)r*rstride + c4);
  }
#endif
  // stage V transposed: Vt[d][key]
#pragma unroll
  for (int j=0;j<8;j++){
    int i = tid + (j<<8);
    int r = i >> 4, c4 = (i & 15) << 2;
    H4 hh; hh.u = *(const uint2*)(vp + (size_t)r*rstride + c4);
#pragma unroll
    for (int q=0;q<4;q++) Vt[(c4+q)*138 + r] = hh.h[q];
  }

  // per-wave Q fragments (rows wave*16..+15, K = 0..63) straight from global
  v16h aq0, aq1;
  {
    const _Float16* qrow = qp + (size_t)((wave<<4) + (lane & 15))*rstride + (((lane>>4)&1)<<3);
#pragma unroll
    for (int i=0;i<8;i++){
      int k = ((i&4)<<2) + ((i&3)<<1);
      aq0[2*i]   = qrow[k];      aq0[2*i+1] = qrow[k+1];
      aq1[2*i]   = qrow[32+k];   aq1[2*i+1] = qrow[32+k+1];
    }
  }
#if HAVE_TDM
  if (tid < 32) tdm_wait0();
#elif HAVE_ASYNC_LDS
  async_wait0();
#endif
  __syncthreads();

  // S = Q K^T : wave's 16x128 score block in 8 accumulators
  v8f acc[8];
#pragma unroll
  for (int t=0;t<8;t++){
    v8f c = zero8();
    c = wmma_f16(aq0, load_BT(Ks, 72, t<<4,  0, lane), c);
    c = wmma_f16(aq1, load_BT(Ks, 72, t<<4, 32, lane), c);
    acc[t] = c;
  }

  // additive mask + row softmax (rows live in 16-lane groups)
  const int hi = lane>>4, nn = lane & 15;
  const float* mrow = mk + (size_t)((wave<<4) + (hi<<3))*128 + nn;
#pragma unroll
  for (int r=0;r<8;r++){
#pragma unroll
    for (int t=0;t<8;t++) acc[t][r] += mrow[(size_t)r*128 + t*16];
    float m = acc[0][r];
#pragma unroll
    for (int t=1;t<8;t++) m = fmaxf(m, acc[t][r]);
    m = redmax16(m);
    float s = 0.f;
#pragma unroll
    for (int t=0;t<8;t++){ float e = __expf(acc[t][r] - m); acc[t][r] = e; s += e; }
    s = redsum16(s);
    float is = 1.f / s;
#pragma unroll
    for (int t=0;t<8;t++) acc[t][r] *= is;
  }

  __syncthreads();   // all waves done reading Ks; reuse region as P

  _Float16* Pw = P + wave*(16*136);
#pragma unroll
  for (int r=0;r<8;r++)
#pragma unroll
    for (int t=0;t<8;t++)
      Pw[(r + (hi<<3))*136 + (t<<4) + nn] = (_Float16)acc[t][r];
  asm volatile("s_wait_dscnt 0" ::: "memory");   // wave-local LDS RAW

  // O = P (16x128) @ V (128x64) ; B[k=key][n=d] = Vt[d][key] (contiguous k)
  v8f oacc[4];
#pragma unroll
  for (int t=0;t<4;t++) oacc[t] = zero8();
#pragma unroll
  for (int ks=0;ks<4;ks++){
    v16h a = load_A(Pw, 136, 0, ks<<5, lane);
#pragma unroll
    for (int t=0;t<4;t++)
      oacc[t] = wmma_f16(a, load_BT(Vt, 138, t<<4, ks<<5, lane), oacc[t]);
  }
#pragma unroll
  for (int t=0;t<4;t++)
#pragma unroll
    for (int r=0;r<8;r++)
      op[(size_t)((wave<<4) + r + (hi<<3))*rstride + (t<<4) + nn] = (_Float16)oacc[t][r];
}

// width-axis attention: slice (b, y, head n); rows are x positions (stride 512)
__global__ __launch_bounds__(256) void csa_attn_w_kernel(
    const _Float16* __restrict__ qh, const _Float16* __restrict__ kh,
    const _Float16* __restrict__ vh, const float* __restrict__ mask,
    _Float16* __restrict__ out)
{
  __shared__ _Float16 smem[17408 + 64*138];
  int s = blockIdx.x;
  int n = s & 7, y = (s >> 3) & 127, b = s >> 10;
  size_t base = ((size_t)(b*128 + y)*128)*512 + n*64;
  csa_attn_slice(qh+base, kh+base, vh+base, mask + (size_t)n*128*128, out+base, 512, smem);
}

// height-axis attention: slice (b, x, head n); rows are y positions (stride 128*512)
__global__ __launch_bounds__(256) void csa_attn_h_kernel(
    const _Float16* __restrict__ qh, const _Float16* __restrict__ kh,
    const _Float16* __restrict__ o1, const float* __restrict__ mask,
    _Float16* __restrict__ out)
{
  __shared__ _Float16 smem[17408 + 64*138];
  int s = blockIdx.x;
  int n = s & 7, x = (s >> 3) & 127, b = s >> 10;
  size_t base = (size_t)b*128*128*512 + (size_t)x*512 + n*64;
  csa_attn_slice(qh+base, kh+base, o1+base, mask + (size_t)n*128*128, out+base, (size_t)128*512, smem);
}

extern "C" void kernel_launch(void* const* d_in, const int* in_sizes, int n_in,
                              void* d_out, int out_size, void* d_ws, size_t ws_size,
                              hipStream_t stream)
{
  (void)in_sizes; (void)n_in; (void)out_size; (void)ws_size;
  const float* x      = (const float*)d_in[0];
  const float* mask_h = (const float*)d_in[1];
  const float* mask_w = (const float*)d_in[2];
  const float* Wq = (const float*)d_in[3];
  const float* bq = (const float*)d_in[4];
  const float* Wk = (const float*)d_in[5];
  const float* bk = (const float*)d_in[6];
  const float* Wv = (const float*)d_in[7];
  const float* bv = (const float*)d_in[8];
  const float* lw = (const float*)d_in[9];
  const float* lb = (const float*)d_in[10];
  const float* Wo = (const float*)d_in[11];
  const float* bo = (const float*)d_in[12];

  const size_t TENS = (size_t)4*128*128*512;   // 33,554,432 elements
  char* p = (char*)d_ws;
  _Float16* qh  = (_Float16*)p; p += TENS*2;
  _Float16* kh  = (_Float16*)p; p += TENS*2;
  _Float16* vh  = (_Float16*)p; p += TENS*2;
  _Float16* o1h = (_Float16*)p; p += TENS*2;
  _Float16* o2h = (_Float16*)p; p += TENS*2;
  _Float16* leh = (_Float16*)p; p += TENS*2;
  _Float16* WqT = (_Float16*)p; p += 512*512*2;
  _Float16* WkT = (_Float16*)p; p += 512*512*2;
  _Float16* WvT = (_Float16*)p; p += 512*512*2;
  _Float16* WoT = (_Float16*)p; p += 512*512*2;

  csa_cvtT_kernel<<<dim3(1024), 256, 0, stream>>>(Wq, WqT);
  csa_cvtT_kernel<<<dim3(1024), 256, 0, stream>>>(Wk, WkT);
  csa_cvtT_kernel<<<dim3(1024), 256, 0, stream>>>(Wv, WvT);
  csa_cvtT_kernel<<<dim3(1024), 256, 0, stream>>>(Wo, WoT);

  csa_proj_gemm_kernel<<<dim3(512,4), 256, 0, stream>>>(x, WqT, bq, 1.0f,   qh);
  csa_proj_gemm_kernel<<<dim3(512,4), 256, 0, stream>>>(x, WkT, bk, 0.125f, kh);  // * KD^-0.5
  csa_proj_gemm_kernel<<<dim3(512,4), 256, 0, stream>>>(x, WvT, bv, 1.0f,   vh);

  csa_lepe_kernel<<<dim3(131072), 256, 0, stream>>>(vh, lw, lb, leh);

  csa_attn_w_kernel<<<dim3(4096), 256, 0, stream>>>(qh, kh, vh,  mask_w, o1h);
  csa_attn_h_kernel<<<dim3(4096), 256, 0, stream>>>(qh, kh, o1h, mask_h, o2h);

  csa_out_gemm_kernel<<<dim3(512,4), 256, 0, stream>>>(o2h, leh, WoT, bo, (float*)d_out);
}
